// PointCloudEncoder_84799834292278
// MI455X (gfx1250) — matrix operand order repaired and checked by reading, not compiled, not run on hardware
//
#include <hip/hip_runtime.h>

// ---------------------------------------------------------------------------
// CDNA5 (gfx1250) wave32 WMMA types
// ---------------------------------------------------------------------------
typedef __attribute__((ext_vector_type(16))) __bf16 v16bf;
typedef __attribute__((ext_vector_type(8)))  __bf16 v8bf;
typedef __attribute__((ext_vector_type(8)))  float  v8f;

#define EPSBN 1e-5f

// ---------------------------------------------------------------------------
// A-fragment loader: 16-bit A 16x32 layout (lanes 0-15: K 0-7,16-23; 16-31: K 8-15,24-31)
// Two 16-byte vector loads per fragment (ds_load_b128 from LDS / b128 global).
// ---------------------------------------------------------------------------
__device__ __forceinline__ v16bf load_a_frag(const __bf16* in, int row, int stride,
                                             int kbase, int lane) {
  const __bf16* p = in + (size_t)row * stride + kbase + ((lane >> 4) << 3);
  v8bf lo = *(const v8bf*)p;
  v8bf hi = *(const v8bf*)(p + 16);
  return __builtin_shufflevector(lo, hi, 0, 1, 2, 3, 4, 5, 6, 7,
                                 8, 9, 10, 11, 12, 13, 14, 15);
}

// ---------------------------------------------------------------------------
// Fold eval-BN into weights, emit bf16 B-fragments + f32 bias.
// frag layout: [ntile][kstep][lane(32)][elem(16)]
// ---------------------------------------------------------------------------
__global__ void fold_weights_kernel(const float* __restrict__ W,
                                    const float* __restrict__ g,
                                    const float* __restrict__ bb,
                                    const float* __restrict__ m,
                                    const float* __restrict__ v,
                                    int cin, int ksteps, int ntiles,
                                    __bf16* __restrict__ frag,
                                    float* __restrict__ bias) {
  int t = blockIdx.x * blockDim.x + threadIdx.x;
  int total = ntiles * ksteps * 512;
  if (t >= total) return;
  int e    = t & 15;
  int lane = (t >> 4) & 31;
  int ks   = (t >> 9) % ksteps;
  int nt   = t / (512 * ksteps);
  int n = nt * 16 + (lane & 15);
  int k = ks * 32 + ((lane >> 4) << 3) + (e & 7) + ((e >> 3) << 4);
  float sc  = g[n] * rsqrtf(v[n] + EPSBN);
  float val = (k < cin) ? W[(size_t)n * cin + k] * sc : 0.0f;
  frag[t] = (__bf16)val;
  if (ks == 0 && e == 0 && lane < 16) bias[n] = bb[n] - m[n] * sc;
}

// ---------------------------------------------------------------------------
// Farthest point sampling: one workgroup (256 thr) per batch, dists in regs,
// LDS argmax tree (first-max-index tie break to match jnp.argmax).
// Writes sampled centers (xyz) directly.   PPT*256 == N.
// ---------------------------------------------------------------------------
template <int PPT>
__global__ void __launch_bounds__(256) fps_kernel(const float* __restrict__ xyz,
                                                  int N, int npoint,
                                                  float* __restrict__ centers) {
  __shared__ float sval[256];
  __shared__ int   sidx[256];
  __shared__ float sc[3];
  const int batch = blockIdx.x;
  const int tid = threadIdx.x;
  const float* X = xyz + (size_t)batch * N * 3;
  float dist[PPT];
#pragma unroll
  for (int j = 0; j < PPT; ++j) dist[j] = 1e10f;
  int cur = 0;  // pointnet2 starts at index 0
  for (int i = 0; i < npoint; ++i) {
    if (tid == 0) {
      float cx = X[cur * 3 + 0], cy = X[cur * 3 + 1], cz = X[cur * 3 + 2];
      sc[0] = cx; sc[1] = cy; sc[2] = cz;
      float* C = centers + ((size_t)batch * npoint + i) * 3;
      C[0] = cx; C[1] = cy; C[2] = cz;
    }
    __syncthreads();
    const float cx = sc[0], cy = sc[1], cz = sc[2];
    float best = -1.0f; int bi = tid;
#pragma unroll
    for (int j = 0; j < PPT; ++j) {
      int p = tid + j * 256;
      float dx = X[p * 3 + 0] - cx;
      float dy = X[p * 3 + 1] - cy;
      float dz = X[p * 3 + 2] - cz;
      float d = dx * dx + dy * dy + dz * dz;
      dist[j] = fminf(dist[j], d);
      if (dist[j] > best) { best = dist[j]; bi = p; }
    }
    sval[tid] = best; sidx[tid] = bi;
    __syncthreads();
    for (int off = 128; off > 0; off >>= 1) {
      if (tid < off) {
        float v2 = sval[tid + off]; int i2 = sidx[tid + off];
        if (v2 > sval[tid] || (v2 == sval[tid] && i2 < sidx[tid])) {
          sval[tid] = v2; sidx[tid] = i2;
        }
      }
      __syncthreads();
    }
    cur = sidx[0];
    __syncthreads();
  }
}

// ---------------------------------------------------------------------------
// Ball query: one wave32 per center; ballot-prefix compaction gives exactly
// the first `nsample` in-radius points in index order; pad with first hit.
// ---------------------------------------------------------------------------
__global__ void __launch_bounds__(256) ball_query_kernel(
    const float* __restrict__ xyz, int N,
    const float* __restrict__ centers, int S, int total, float r2,
    int* __restrict__ nidx) {
  int wid = (blockIdx.x * blockDim.x + threadIdx.x) >> 5;
  int lane = threadIdx.x & 31;
  if (wid >= total) return;
  int batch = wid / S;
  const float* X = xyz + (size_t)batch * N * 3;
  const float* C = centers + (size_t)wid * 3;
  const float cx = C[0], cy = C[1], cz = C[2];
  int* out = nidx + (size_t)wid * 64;
  int count = 0, first = -1;
  for (int base = 0; base < N && count < 64; base += 32) {
    int p = base + lane;
    float dx = X[p * 3 + 0] - cx;
    float dy = X[p * 3 + 1] - cy;
    float dz = X[p * 3 + 2] - cz;
    bool hit = (dx * dx + dy * dy + dz * dz) < r2;
    unsigned mask = (unsigned)__ballot(hit);
    if (first < 0 && mask) first = base + __ffs(mask) - 1;
    int slot = count + __popc(mask & ((1u << lane) - 1u));
    if (hit && slot < 64) out[slot] = p;
    count += __popc(mask);
  }
  if (count > 64) count = 64;
  if (first < 0) first = 0;
  for (int slot = count + lane; slot < 64; slot += 32) out[slot] = first;
}

// ---------------------------------------------------------------------------
// One 16xCOUT WMMA MLP layer for a 64-row group, 4 waves = 4 M-tiles.
// Register-blocked 4 N-tiles per A-fragment load (COUT multiple of 64).
// in/out in LDS (row-major, stride SIN/SOUT bf16), weights pre-fragmented.
// ---------------------------------------------------------------------------
template <int KP, int COUT, int SIN, int SOUT>
__device__ __forceinline__ void mlp_layer(const __bf16* in, __bf16* out,
                                          const __bf16* __restrict__ frag,
                                          const float* __restrict__ bias) {
  const int lane  = threadIdx.x & 31;
  const int mt    = threadIdx.x >> 5;  // wave = M tile
  const int KS    = KP / 32;
  const int NT    = COUT / 16;
  const int ncol  = lane & 15;
  const int mhalf = (lane >> 4) * 8;
  const int arow  = mt * 16 + (lane & 15);
#pragma unroll
  for (int ntb = 0; ntb < NT; ntb += 4) {
    v8f acc[4];
#pragma unroll
    for (int j = 0; j < 4; ++j)
      acc[j] = (v8f){0.f, 0.f, 0.f, 0.f, 0.f, 0.f, 0.f, 0.f};
#pragma unroll
    for (int ks = 0; ks < KS; ++ks) {
      v16bf a = load_a_frag(in, arow, SIN, ks * 32, lane);
#pragma unroll
      for (int j = 0; j < 4; ++j) {
        v16bf b = *(const v16bf*)(frag +
            (((size_t)(ntb + j) * KS + ks) * 32 + lane) * 16);
        acc[j] = __builtin_amdgcn_wmma_f32_16x16x32_bf16(
            false, a, false, b, (short)0, acc[j], false, false);
      }
    }
#pragma unroll
    for (int j = 0; j < 4; ++j) {
      const float bv = bias[(ntb + j) * 16 + ncol];
#pragma unroll
      for (int e = 0; e < 8; ++e) {
        float x = acc[j][e] + bv;
        x = fmaxf(x, 0.0f);  // BN folded, ReLU
        out[(size_t)(mt * 16 + mhalf + e) * SOUT + (ntb + j) * 16 + ncol] =
            (__bf16)x;
      }
    }
  }
}

// ---------------------------------------------------------------------------
// Fused SA group kernel: gather(center-relative xyz ++ feats) -> 3-layer
// WMMA MLP in LDS -> max over 64 neighbors.  One block (128 thr) per group.
// ---------------------------------------------------------------------------
template <int CFEAT, int KP0, int C1, int C2, int C3, int SA, int SB>
__global__ void __launch_bounds__(128) sa_group_kernel(
    const float* __restrict__ xyz, int Nx,
    const float* __restrict__ feats,
    const float* __restrict__ centers,
    const int* __restrict__ nidx, int S,
    const __bf16* __restrict__ f1, const float* __restrict__ b1,
    const __bf16* __restrict__ f2, const float* __restrict__ b2,
    const __bf16* __restrict__ f3, const float* __restrict__ b3,
    float* __restrict__ out) {
  __shared__ __bf16 bufA[64 * SA];
  __shared__ __bf16 bufB[64 * SB];
  const int tid = threadIdx.x;
  const int grp = blockIdx.x;
  const int batch = grp / S;
  // zero bufA (covers K-padding of the first layer)
  for (int i = tid; i < 64 * SA; i += 128) bufA[i] = (__bf16)0.0f;
  __syncthreads();
  // gather the 64-neighbor group
  const float* C = centers + (size_t)grp * 3;
  const float cx = C[0], cy = C[1], cz = C[2];
  const int* NI = nidx + (size_t)grp * 64;
  if (tid < 64) {
    int p = NI[tid];
    const float* P = xyz + ((size_t)batch * Nx + p) * 3;
    __bf16* row = bufA + (size_t)tid * SA;
    row[0] = (__bf16)(P[0] - cx);
    row[1] = (__bf16)(P[1] - cy);
    row[2] = (__bf16)(P[2] - cz);
    const float* F = feats + ((size_t)batch * Nx + p) * CFEAT;
    for (int c = 0; c < CFEAT; ++c) row[3 + c] = (__bf16)F[c];
  }
  __syncthreads();
  mlp_layer<KP0, C1, SA, SB>(bufA, bufB, f1, b1);
  __syncthreads();
  mlp_layer<C1, C2, SB, SA>(bufB, bufA, f2, b2);
  __syncthreads();
  mlp_layer<C2, C3, SA, SB>(bufA, bufB, f3, b3);
  __syncthreads();
  // max over the 64 neighbor rows
  float* O = out + (size_t)grp * C3;
  for (int c = tid; c < C3; c += 128) {
    float mx = -1e30f;
    for (int r = 0; r < 64; ++r) mx = fmaxf(mx, (float)bufB[(size_t)r * SB + c]);
    O[c] = mx;
  }
}

// ---------------------------------------------------------------------------
// Generic fragment-weight WMMA GEMM + bias + ReLU (A bf16 row-major global).
// One wave per 16x64 output strip (4 N-tiles register-blocked).
// ---------------------------------------------------------------------------
__global__ void __launch_bounds__(128) gemm_frag_kernel(
    const __bf16* __restrict__ A, int lda,
    const __bf16* __restrict__ frag, const float* __restrict__ bias,
    __bf16* __restrict__ out, int ldo,
    int Mtiles, int Ntiles, int Ksteps, int relu) {
  const int lane = threadIdx.x & 31;
  const int nt4 = Ntiles >> 2;
  const int t = blockIdx.x * 4 + (threadIdx.x >> 5);
  if (t >= Mtiles * nt4) return;
  const int mt = t / nt4;
  const int nt0 = (t % nt4) * 4;
  const int arow = mt * 16 + (lane & 15);
  const int ncol = lane & 15;
  const int mhalf = (lane >> 4) * 8;
  v8f acc[4];
#pragma unroll
  for (int j = 0; j < 4; ++j)
    acc[j] = (v8f){0.f, 0.f, 0.f, 0.f, 0.f, 0.f, 0.f, 0.f};
  for (int ks = 0; ks < Ksteps; ++ks) {
    v16bf a = load_a_frag(A, arow, lda, ks * 32, lane);
#pragma unroll
    for (int j = 0; j < 4; ++j) {
      v16bf b = *(const v16bf*)(frag +
          (((size_t)(nt0 + j) * Ksteps + ks) * 32 + lane) * 16);
      acc[j] = __builtin_amdgcn_wmma_f32_16x16x32_bf16(
          false, a, false, b, (short)0, acc[j], false, false);
    }
  }
#pragma unroll
  for (int j = 0; j < 4; ++j) {
    const float bv = bias[(nt0 + j) * 16 + ncol];
#pragma unroll
    for (int e = 0; e < 8; ++e) {
      float x = acc[j][e] + bv;
      if (relu) x = fmaxf(x, 0.0f);
      out[(size_t)(mt * 16 + mhalf + e) * ldo + (nt0 + j) * 16 + ncol] =
          (__bf16)x;
    }
  }
}

// ---------------------------------------------------------------------------
// Build sa3 input rows: [xyz2(3) ++ f2(256) ++ zero-pad -> 288] bf16
// ---------------------------------------------------------------------------
__global__ void build_x0_kernel(const float* __restrict__ xyz2,
                                const float* __restrict__ f2,
                                __bf16* __restrict__ X0, int rows) {
  int idx = blockIdx.x * blockDim.x + threadIdx.x;
  if (idx >= rows * 288) return;
  int r = idx / 288, c = idx % 288;
  float v = 0.0f;
  if (c < 3)        v = xyz2[(size_t)r * 3 + c];
  else if (c < 259) v = f2[(size_t)r * 256 + (c - 3)];
  X0[idx] = (__bf16)v;
}

// ---------------------------------------------------------------------------
// Max over `rows` group rows per (batch, channel)
// ---------------------------------------------------------------------------
__global__ void maxpool_rows_kernel(const __bf16* __restrict__ Y, int rows,
                                    int C, float* __restrict__ g, int total) {
  int idx = blockIdx.x * blockDim.x + threadIdx.x;
  if (idx >= total) return;
  int b = idx / C, c = idx % C;
  const __bf16* p = Y + (size_t)b * rows * C + c;
  float mx = -1e30f;
  for (int r = 0; r < rows; ++r) mx = fmaxf(mx, (float)p[(size_t)r * C]);
  g[idx] = mx;
}

// ---------------------------------------------------------------------------
// Final head: h = g @ W + b ; LayerNorm(256) ; exact GELU.  One block/batch.
// ---------------------------------------------------------------------------
__global__ void __launch_bounds__(256) proj_ln_gelu_kernel(
    const float* __restrict__ g, const float* __restrict__ W,
    const float* __restrict__ pb, const float* __restrict__ lng,
    const float* __restrict__ lnb, float* __restrict__ out) {
  __shared__ float red[256];
  const int b = blockIdx.x, o = threadIdx.x;
  const float* G = g + (size_t)b * 1024;
  float acc = 0.0f;
  for (int k = 0; k < 1024; ++k) acc = fmaf(G[k], W[(size_t)k * 256 + o], acc);
  const float h = acc + pb[o];
  red[o] = h;
  __syncthreads();
  for (int off = 128; off > 0; off >>= 1) {
    if (o < off) red[o] += red[o + off];
    __syncthreads();
  }
  const float mu = red[0] * (1.0f / 256.0f);
  __syncthreads();
  const float d = h - mu;
  red[o] = d * d;
  __syncthreads();
  for (int off = 128; off > 0; off >>= 1) {
    if (o < off) red[o] += red[o + off];
    __syncthreads();
  }
  const float var = red[0] * (1.0f / 256.0f);
  float xn = d * rsqrtf(var + EPSBN) * lng[o] + lnb[o];
  out[(size_t)b * 256 + o] = 0.5f * xn * (1.0f + erff(xn * 0.70710678118654752f));
}

// ---------------------------------------------------------------------------
// Host-side launch
// ---------------------------------------------------------------------------
extern "C" void kernel_launch(void* const* d_in, const int* in_sizes, int n_in,
                              void* d_out, int out_size, void* d_ws, size_t ws_size,
                              hipStream_t stream) {
  (void)in_sizes; (void)n_in; (void)out_size; (void)ws_size;

  const float* pc = (const float*)d_in[0];  // [16,4096,3]
  const float *LW[9], *Lg[9], *Lb[9], *Lm[9], *Lv[9];
  for (int l = 0; l < 9; ++l) {
    LW[l] = (const float*)d_in[1 + l * 5 + 0];
    Lg[l] = (const float*)d_in[1 + l * 5 + 1];
    Lb[l] = (const float*)d_in[1 + l * 5 + 2];
    Lm[l] = (const float*)d_in[1 + l * 5 + 3];
    Lv[l] = (const float*)d_in[1 + l * 5 + 4];
  }
  const float* projW = (const float*)d_in[46];  // [1024,256]
  const float* projB = (const float*)d_in[47];
  const float* lnG   = (const float*)d_in[48];
  const float* lnB   = (const float*)d_in[49];

  static const int cin9[9]  = {6, 64, 64, 131, 128, 128, 259, 256, 512};
  static const int cout9[9] = {64, 64, 128, 128, 128, 256, 256, 512, 1024};

  char* ws = (char*)d_ws;
  size_t off = 0;
  auto alloc = [&](size_t n) -> void* {
    off = (off + 255) & ~(size_t)255;
    void* p = (void*)(ws + off);
    off += n;
    return p;
  };

  __bf16* frag[9]; float* bias[9];
  int ksteps[9], ntiles[9];
  for (int l = 0; l < 9; ++l) {
    int kp = (cin9[l] + 31) / 32 * 32;
    ksteps[l] = kp / 32;
    ntiles[l] = cout9[l] / 16;
    frag[l] = (__bf16*)alloc((size_t)ntiles[l] * ksteps[l] * 512 * sizeof(__bf16));
    bias[l] = (float*)alloc((size_t)cout9[l] * sizeof(float));
  }
  float* xyz1  = (float*)alloc((size_t)16 * 512 * 3 * 4);
  int*   nidx1 = (int*)  alloc((size_t)16 * 512 * 64 * 4);
  float* fo1   = (float*)alloc((size_t)16 * 512 * 128 * 4);
  float* xyz2  = (float*)alloc((size_t)16 * 128 * 3 * 4);
  int*   nidx2 = (int*)  alloc((size_t)16 * 128 * 64 * 4);
  float* fo2   = (float*)alloc((size_t)16 * 128 * 256 * 4);
  __bf16* X0   = (__bf16*)alloc((size_t)2048 * 288 * 2);
  __bf16* Y1   = (__bf16*)alloc((size_t)2048 * 256 * 2);
  __bf16* Y2   = (__bf16*)alloc((size_t)2048 * 512 * 2);
  __bf16* Y3   = (__bf16*)alloc((size_t)2048 * 1024 * 2);
  float*  gbuf = (float*)alloc((size_t)16 * 1024 * 4);

  // 0) fold BN into bf16 WMMA weight fragments
  for (int l = 0; l < 9; ++l) {
    int tot = ntiles[l] * ksteps[l] * 512;
    fold_weights_kernel<<<(tot + 255) / 256, 256, 0, stream>>>(
        LW[l], Lg[l], Lb[l], Lm[l], Lv[l], cin9[l], ksteps[l], ntiles[l],
        frag[l], bias[l]);
  }

  // 1) SA1: FPS(4096->512), ball query r=0.2, fused group MLP [6,64,64,128]
  fps_kernel<16><<<16, 256, 0, stream>>>(pc, 4096, 512, xyz1);
  ball_query_kernel<<<(16 * 512) / 8, 256, 0, stream>>>(
      pc, 4096, xyz1, 512, 16 * 512, 0.04f, nidx1);
  sa_group_kernel<3, 32, 64, 64, 128, 64, 128><<<16 * 512, 128, 0, stream>>>(
      pc, 4096, pc, xyz1, nidx1, 512,
      frag[0], bias[0], frag[1], bias[1], frag[2], bias[2], fo1);

  // 2) SA2: FPS(512->128), ball query r=0.4, fused group MLP [131,128,128,256]
  fps_kernel<2><<<16, 256, 0, stream>>>(xyz1, 512, 128, xyz2);
  ball_query_kernel<<<(16 * 128) / 8, 256, 0, stream>>>(
      xyz1, 512, xyz2, 128, 16 * 128, 0.16f, nidx2);
  sa_group_kernel<128, 160, 128, 128, 256, 160, 256><<<16 * 128, 128, 0, stream>>>(
      xyz1, 512, fo1, xyz2, nidx2, 128,
      frag[3], bias[3], frag[4], bias[4], frag[5], bias[5], fo2);

  // 3) SA3 global MLP [259,256,512,1024] as fragment-weight WMMA GEMMs
  //    tiles per launch = Mtiles * (Ntiles/4), one wave per 16x64 strip
  build_x0_kernel<<<(2048 * 288 + 255) / 256, 256, 0, stream>>>(xyz2, fo2, X0, 2048);
  gemm_frag_kernel<<<(128 * 4 + 3) / 4, 128, 0, stream>>>(
      X0, 288, frag[6], bias[6], Y1, 256, 128, 16, 9, 1);
  gemm_frag_kernel<<<(128 * 8 + 3) / 4, 128, 0, stream>>>(
      Y1, 256, frag[7], bias[7], Y2, 512, 128, 32, 8, 1);
  gemm_frag_kernel<<<(128 * 16 + 3) / 4, 128, 0, stream>>>(
      Y2, 512, frag[8], bias[8], Y3, 1024, 128, 64, 16, 1);
  maxpool_rows_kernel<<<(16 * 1024 + 255) / 256, 256, 0, stream>>>(
      Y3, 128, 1024, gbuf, 16 * 1024);

  // 4) projection + LayerNorm + exact GELU
  proj_ln_gelu_kernel<<<16, 256, 0, stream>>>(gbuf, projW, projB, lnG, lnB,
                                              (float*)d_out);
}
